// GCE_TAGNN_v2_58067957842017
// MI455X (gfx1250) — compile-verified, matrix operand order checked by reading, not run on hardware
//
#include <hip/hip_runtime.h>
#include <hip/hip_bf16.h>
#include <math.h>

#define NITEMS 10000
#define HDIM   128
#define SLEN   50
#define BATCH  64
#define NHEAD  8
#define NEDGE  160000
#define NCAND  9999

typedef __bf16 bf16_t;
typedef __attribute__((ext_vector_type(16))) __bf16 v16bf;
typedef __attribute__((ext_vector_type(8)))  __bf16 v8bf;
typedef __attribute__((ext_vector_type(8)))  float  v8f;

// ---------------------------------------------------------------------------
// WMMA helpers (wave32, V_WMMA_F32_16X16X32_BF16)
// 16-bit A (16x32 MxK) fragment layout per CDNA5 ISA:
//   lane<16  (m=lane)    : e[0..7]=A[m,kb+0..7],  e[8..15]=A[m,kb+16..23]
//   lane>=16 (m=lane-16) : e[0..7]=A[m,kb+8..15], e[8..15]=A[m,kb+24..31]
// B (32x16 KxN) is the mirror under transpose, so loading B from a row-major
// (N,K) "weight" matrix uses the identical per-lane gather -> both operands
// come from two contiguous 16-byte loads.
// ---------------------------------------------------------------------------
__device__ __forceinline__ v16bf frag_load(const bf16_t* p, int ld, int row0,
                                           int rowmax, int kb)
{
    int lane = threadIdx.x & 31;
    int r = row0 + (lane & 15);
    if (r > rowmax) r = rowmax;                 // clamp (edge tiles)
    int hi = lane >> 4;
    const bf16_t* base = p + (size_t)r * ld + kb + hi * 8;
    v8bf lo = *(const v8bf*)base;
    v8bf hv = *(const v8bf*)(base + 16);
    return __builtin_shufflevector(lo, hv, 0,1,2,3,4,5,6,7,
                                           8,9,10,11,12,13,14,15);
}

__device__ __forceinline__ v8f wmma_bf16(v16bf a, v16bf b, v8f c)
{
    return __builtin_amdgcn_wmma_f32_16x16x32_bf16(false, a, false, b,
                                                   (short)0, c, false, false);
}

// Async global->LDS 16B copy (CDNA5 GLOBAL_LOAD_ASYNC_TO_LDS_B128, GV mode).
// lds_off = LDS byte offset (low 32 bits of the generic shared address, per
// the LDS aperture mapping in CDNA5 ISA). Tracked by
// ASYNCcnt; completion enforced with s_wait_asynccnt before the barrier.
__device__ __forceinline__ void async_copy16(unsigned lds_off, const void* g)
{
    asm volatile("global_load_async_to_lds_b128 %0, %1, off"
                 :: "v"(lds_off), "v"(g) : "memory");
}

__device__ __forceinline__ void async_wait0()
{
    asm volatile("s_wait_asynccnt 0x0" ::: "memory");
}

// ---------------------------------------------------------------------------
// Generic GEMM: C[M,N] = act(A[M,K] @ W[N,K]^T + bias), bf16 in, f32 acc.
// block = 128 threads (4 waves); wave w owns n-tile blockIdx.y*4+w.
// ---------------------------------------------------------------------------
__global__ void k_gemm(const bf16_t* __restrict__ A, int lda,
                       const bf16_t* __restrict__ W, int ldw,
                       const float*  __restrict__ bias,
                       float* Cf, bf16_t* Cb, int ldc,
                       int M, int N, int K, int relu)
{
    int wave = threadIdx.x >> 5;
    int n0 = (blockIdx.y * 4 + wave) * 16;
    int m0 = blockIdx.x * 16;
    if (n0 >= N || m0 >= M) return;            // wave-uniform exit
    v8f acc = {};
    for (int kb = 0; kb < K; kb += 32) {
        v16bf a = frag_load(A, lda, m0, M - 1, kb);
        v16bf b = frag_load(W, ldw, n0, N - 1, kb);
        acc = wmma_bf16(a, b, acc);
    }
    int lane = threadIdx.x & 31;
    int col = lane & 15, hi = lane >> 4;
    int cc = n0 + col;
    float bv = (bias && cc < N) ? bias[cc] : 0.0f;
#pragma unroll
    for (int i = 0; i < 8; ++i) {
        int rr = m0 + i + 8 * hi;
        if (rr < M && cc < N) {
            float v = acc[i] + bv;
            if (relu) v = fmaxf(v, 0.0f);
            if (Cf) Cf[(size_t)rr * ldc + cc] = v;
            if (Cb) Cb[(size_t)rr * ldc + cc] = (bf16_t)v;
        }
    }
}

// ---------------------------------------------------------------------------
// Small utility kernels
// ---------------------------------------------------------------------------
__global__ void k_zero(float* p, int n)
{
    int i = blockIdx.x * 256 + threadIdx.x;
    if (i < n) p[i] = 0.0f;
}

__global__ void k_cvt(const float* __restrict__ s, bf16_t* __restrict__ d, int n)
{
    int i = blockIdx.x * 256 + threadIdx.x;
    if (i < n) d[i] = (bf16_t)s[i];
}

// d[c*rows + r] = s[r*cols + c]  (f32 -> bf16 transpose, for w3_W)
__global__ void k_tcvt(const float* __restrict__ s, bf16_t* __restrict__ d,
                       int rows, int cols)
{
    int i = blockIdx.x * 256 + threadIdx.x;
    if (i < rows * cols) {
        int r = i / cols, c = i % cols;
        d[(size_t)c * rows + r] = (bf16_t)s[i];
    }
}

// agg[row[e],:] += w[e] * emb[col[e],:]   (global graph message passing)
__global__ void k_scatter(const int* __restrict__ row, const int* __restrict__ col,
                          const float* __restrict__ w, const float* __restrict__ emb,
                          float* agg)
{
    int e = blockIdx.x, h = threadIdx.x;
    float v = w[e] * emb[(size_t)col[e] * HDIM + h];
    atomicAdd(&agg[(size_t)row[e] * HDIM + h], v);
}

// h0 = emb[items] (f32 + bf16), sess_glob = hg[items]
__global__ void k_gather(const int* __restrict__ items, const float* __restrict__ emb,
                         const float* __restrict__ hg, float* h0f, bf16_t* h0b,
                         float* sgf)
{
    int r = blockIdx.x, h = threadIdx.x;
    int it = items[r];
    float e = emb[(size_t)it * HDIM + h];
    h0f[(size_t)r * HDIM + h] = e;
    h0b[(size_t)r * HDIM + h] = (bf16_t)e;
    sgf[(size_t)r * HDIM + h] = hg[(size_t)it * HDIM + h];
}

// ---------------------------------------------------------------------------
// Batched inp = [adj@hin , adj@hout]  per session (pad 50 -> 64, K pad -> 64)
// block = 256 threads (8 waves), one session per block, WMMA from LDS.
// ---------------------------------------------------------------------------
__global__ void k_adjmm(const float* __restrict__ adj, const bf16_t* __restrict__ hin,
                        const bf16_t* __restrict__ hout, bf16_t* __restrict__ inp)
{
    __shared__ __align__(16) bf16_t sA[64 * 64];     // adj padded
    __shared__ __align__(16) bf16_t sBT[256 * 64];   // (N=256, K=64) layout
    int b = blockIdx.x;
    for (int idx = threadIdx.x; idx < 64 * 64; idx += 256) {
        int l = idx >> 6, k = idx & 63;
        float v = (l < SLEN && k < SLEN) ? adj[((size_t)b * SLEN + l) * SLEN + k] : 0.0f;
        sA[idx] = (bf16_t)v;
    }
    for (int idx = threadIdx.x; idx < 256 * 64; idx += 256) {
        int n = idx >> 6, k = idx & 63;
        bf16_t v = (bf16_t)0.0f;
        if (k < SLEN)
            v = (n < HDIM) ? hin [((size_t)b * SLEN + k) * HDIM + n]
                           : hout[((size_t)b * SLEN + k) * HDIM + (n - HDIM)];
        sBT[idx] = v;
    }
    __syncthreads();
    int wave = threadIdx.x >> 5;
    int lane = threadIdx.x & 31;
    int col = lane & 15, hi = lane >> 4;
#pragma unroll
    for (int t = 0; t < 2; ++t) {
        int n0 = (wave + 8 * t) * 16;
#pragma unroll
        for (int mt = 0; mt < 4; ++mt) {
            int m0 = mt * 16;
            v8f acc = {};
#pragma unroll
            for (int kb = 0; kb < 64; kb += 32) {
                v16bf a  = frag_load(sA, 64, m0, 63, kb);
                v16bf bb = frag_load(sBT, 64, n0, 255, kb);
                acc = wmma_bf16(a, bb, acc);
            }
#pragma unroll
            for (int i = 0; i < 8; ++i) {
                int l = m0 + i + 8 * hi;
                if (l < SLEN)
                    inp[((size_t)b * SLEN + l) * 256 + n0 + col] = (bf16_t)acc[i];
            }
        }
    }
}

// ---------------------------------------------------------------------------
// GRU update + positional embedding + last-row extraction (fp32 VALU)
// ---------------------------------------------------------------------------
__global__ void k_gru_final(const float* __restrict__ gi, const float* __restrict__ gh,
                            const float* __restrict__ h0, const float* __restrict__ sg,
                            const float* __restrict__ pos, const int* __restrict__ items,
                            const int* __restrict__ lens, bf16_t* finb, bf16_t* lastsg)
{
    int row = blockIdx.x, h = threadIdx.x;
    int b = row / SLEN, l = row % SLEN;
    size_t r3 = (size_t)row * 384;
    float ir = gi[r3 + h], ii = gi[r3 + 128 + h], in_ = gi[r3 + 256 + h];
    float hr = gh[r3 + h], hi_ = gh[r3 + 128 + h], hn = gh[r3 + 256 + h];
    float r = 1.0f / (1.0f + expf(-(ir + hr)));
    float z = 1.0f / (1.0f + expf(-(ii + hi_)));
    float n = tanhf(in_ + r * hn);
    float h0v = h0[(size_t)row * HDIM + h];
    float hnew = n + z * (h0v - n);
    float rich = sg[(size_t)row * HDIM + h] + hnew;
    int it = items[row], len = lens[b];
    int rev = (it == 0) ? 0 : (len - 1 - l);
    float fin = rich + pos[(size_t)rev * HDIM + h];
    finb[(size_t)row * HDIM + h] = (bf16_t)fin;
    if (l == len - 1) lastsg[(size_t)b * 256 + h] = (bf16_t)fin;
}

// ---------------------------------------------------------------------------
// Multi-head attention over L=50 (tiny: 512 head-units, hd=16) -> ctx (bf16)
// ---------------------------------------------------------------------------
__global__ void k_attn(const float* __restrict__ q, const float* __restrict__ kv,
                       const int* __restrict__ items, bf16_t* ctxb)
{
    __shared__ float sred[64];
    __shared__ float sexp[64];
    int b = blockIdx.x >> 3, hh = blockIdx.x & 7;
    int t = threadIdx.x;
    float s = -1e30f;
    if (t < SLEN && items[b * SLEN + t] != 0) {
        float acc = 0.0f;
        for (int d = 0; d < 16; ++d)
            acc += q[(size_t)b * HDIM + hh * 16 + d] *
                   kv[((size_t)b * SLEN + t) * 256 + hh * 16 + d];
        s = acc * 0.25f;   // 1/sqrt(16)
    }
    sred[t] = s; __syncthreads();
    for (int off = 32; off > 0; off >>= 1) {
        if (t < off) sred[t] = fmaxf(sred[t], sred[t + off]);
        __syncthreads();
    }
    float mx = sred[0]; __syncthreads();
    float e = (s <= -1e29f) ? 0.0f : expf(s - mx);
    sexp[t] = e; sred[t] = e; __syncthreads();
    for (int off = 32; off > 0; off >>= 1) {
        if (t < off) sred[t] += sred[t + off];
        __syncthreads();
    }
    float den = sred[0];
    if (t < 16) {
        float acc = 0.0f;
        for (int l = 0; l < SLEN; ++l)
            acc += sexp[l] * kv[((size_t)b * SLEN + l) * 256 + HDIM + hh * 16 + t];
        ctxb[(size_t)b * HDIM + hh * 16 + t] = (bf16_t)(acc / den);
    }
}

// ---------------------------------------------------------------------------
// Fused target-attention scoring (the hot kernel):
// per (b, 16-wide n-tile): tsc = fw[b] @ cand^T, g = final[b] @ u1^T (WMMA),
// masked softmax over L, scores = <talpha,g> + extra[b,n].
// LDS staging of fw[b]/final[b] uses the CDNA5 async global->LDS engine.
// block = 128 threads (4 waves, wave = m-tile of padded-64 L dim).
// ---------------------------------------------------------------------------
__global__ void k_score(const bf16_t* __restrict__ fwb, const bf16_t* __restrict__ finb,
                        const bf16_t* __restrict__ candb, const bf16_t* __restrict__ ub,
                        const float* __restrict__ extra, const int* __restrict__ items,
                        float* __restrict__ out)
{
    __shared__ __align__(16) bf16_t sFW[64 * HDIM];
    __shared__ __align__(16) bf16_t sFI[64 * HDIM];
    __shared__ float sT[64 * 16];
    __shared__ float sG[64 * 16];
    __shared__ int   sMask[64];
    int b = blockIdx.y;
    int n0 = blockIdx.x * 16;

    // LDS byte offsets (low 32 bits of generic shared address = LDS aperture)
    unsigned fw_lds = (unsigned)(size_t)(void*)sFW;
    unsigned fi_lds = (unsigned)(size_t)(void*)sFI;

    // async global->LDS copy of the 50 valid rows, 16B chunks (ASYNCcnt)
    for (int idx = threadIdx.x; idx < SLEN * (HDIM / 8); idx += 128) {
        int r = idx >> 4, ch = idx & 15;            // 16 chunks per row
        unsigned loff = (unsigned)((r * HDIM + ch * 8) * 2);
        size_t   goff = ((size_t)b * SLEN + r) * HDIM + ch * 8;
        async_copy16(fw_lds + loff, fwb + goff);
        async_copy16(fi_lds + loff, finb + goff);
    }
    // zero-fill padding rows [50,64) with plain LDS stores
    for (int idx = threadIdx.x; idx < (64 - SLEN) * 16; idx += 128) {
        int r = SLEN + (idx >> 4), ch = idx & 15;
        uint4 z = {0, 0, 0, 0};
        ((uint4*)sFW)[r * 16 + ch] = z;
        ((uint4*)sFI)[r * 16 + ch] = z;
    }
    if (threadIdx.x < 64)
        sMask[threadIdx.x] = (threadIdx.x < SLEN) ? items[b * SLEN + threadIdx.x] : 0;
    async_wait0();          // drain ASYNCcnt before making LDS visible
    __syncthreads();

    int wave = threadIdx.x >> 5;
    int m0 = wave * 16;
    v8f accT = {}, accG = {};
#pragma unroll
    for (int kb = 0; kb < HDIM; kb += 32) {
        v16bf aF = frag_load(sFW, HDIM, m0, 63, kb);
        v16bf bC = frag_load(candb, HDIM, n0, NCAND - 1, kb);
        accT = wmma_bf16(aF, bC, accT);
        v16bf aI = frag_load(sFI, HDIM, m0, 63, kb);
        v16bf bU = frag_load(ub, 384, n0, NCAND - 1, kb);
        accG = wmma_bf16(aI, bU, accG);
    }
    int lane = threadIdx.x & 31;
    int col = lane & 15, hi = lane >> 4;
#pragma unroll
    for (int i = 0; i < 8; ++i) {
        int l = m0 + i + 8 * hi;
        sT[l * 16 + col] = accT[i];
        sG[l * 16 + col] = accG[i];
    }
    __syncthreads();

    if (threadIdx.x < 16) {
        int n = n0 + threadIdx.x;
        if (n < NCAND) {
            float mx = -1e30f;
            for (int l = 0; l < SLEN; ++l)
                if (sMask[l] != 0) mx = fmaxf(mx, sT[l * 16 + threadIdx.x]);
            float den = 0.0f, num = 0.0f;
            for (int l = 0; l < SLEN; ++l)
                if (sMask[l] != 0) {
                    float e = expf(sT[l * 16 + threadIdx.x] - mx);
                    den += e;
                    num += e * sG[l * 16 + threadIdx.x];
                }
            out[(size_t)b * NCAND + n] = num / den + extra[(size_t)b * NCAND + n];
        }
    }
}

// ---------------------------------------------------------------------------
// Host-side orchestration
// ---------------------------------------------------------------------------
static inline void gemm(hipStream_t s, const bf16_t* A, int lda,
                        const bf16_t* W, int ldw, const float* bias,
                        float* Cf, bf16_t* Cb, int ldc,
                        int M, int N, int K, int relu)
{
    dim3 g((M + 15) / 16, (N + 63) / 64);
    k_gemm<<<g, 128, 0, s>>>(A, lda, W, ldw, bias, Cf, Cb, ldc, M, N, K, relu);
}

extern "C" void kernel_launch(void* const* d_in, const int* in_sizes, int n_in,
                              void* d_out, int out_size, void* d_ws, size_t ws_size,
                              hipStream_t stream)
{
    (void)in_sizes; (void)n_in; (void)out_size; (void)ws_size;
    const int*   items  = (const int*)d_in[0];
    const int*   lens   = (const int*)d_in[1];
    const float* adj    = (const float*)d_in[2];
    const int*   erow   = (const int*)d_in[3];
    const int*   ecol   = (const int*)d_in[4];
    const float* ew     = (const float*)d_in[5];
    const float* emb    = (const float*)d_in[6];
    const float* pos    = (const float*)d_in[7];
    const float* gW     = (const float*)d_in[8];
    const float* gb     = (const float*)d_in[9];
    const float* liW    = (const float*)d_in[10];
    const float* liB    = (const float*)d_in[11];
    const float* loW    = (const float*)d_in[12];
    const float* loB    = (const float*)d_in[13];
    const float* wih    = (const float*)d_in[14];
    const float* whh    = (const float*)d_in[15];
    const float* bih    = (const float*)d_in[16];
    const float* bhh    = (const float*)d_in[17];
    const float* inW    = (const float*)d_in[18];
    const float* inB    = (const float*)d_in[19];
    const float* outW   = (const float*)d_in[20];
    const float* outB   = (const float*)d_in[21];
    const float* wtW    = (const float*)d_in[22];
    const float* w3     = (const float*)d_in[23];
    float* out = (float*)d_out;

    char* p = (char*)d_ws;
    auto alloc = [&](size_t n) { char* r = p; p += (n + 255) & ~((size_t)255); return r; };

    float*  aggf   = (float*) alloc((size_t)NITEMS * HDIM * 4);
    bf16_t* aggb   = (bf16_t*)alloc((size_t)NITEMS * HDIM * 2);
    float*  hgf    = (float*) alloc((size_t)NITEMS * HDIM * 4);
    bf16_t* embb   = (bf16_t*)alloc((size_t)NITEMS * HDIM * 2);
    bf16_t* gWb    = (bf16_t*)alloc(HDIM * HDIM * 2);
    bf16_t* liWb   = (bf16_t*)alloc(HDIM * HDIM * 2);
    bf16_t* loWb   = (bf16_t*)alloc(HDIM * HDIM * 2);
    bf16_t* wihb   = (bf16_t*)alloc(384 * 256 * 2);
    bf16_t* whhb   = (bf16_t*)alloc(384 * 128 * 2);
    bf16_t* inWb   = (bf16_t*)alloc(384 * 128 * 2);
    bf16_t* outWb  = (bf16_t*)alloc(HDIM * HDIM * 2);
    bf16_t* wtWb   = (bf16_t*)alloc(HDIM * HDIM * 2);
    bf16_t* w3tb   = (bf16_t*)alloc(384 * 128 * 2);
    float*  h0f    = (float*) alloc((size_t)BATCH * SLEN * HDIM * 4);
    bf16_t* h0b    = (bf16_t*)alloc((size_t)BATCH * SLEN * HDIM * 2);
    float*  sgfb   = (float*) alloc((size_t)BATCH * SLEN * HDIM * 4);
    bf16_t* hinb   = (bf16_t*)alloc((size_t)BATCH * SLEN * HDIM * 2);
    bf16_t* houtb  = (bf16_t*)alloc((size_t)BATCH * SLEN * HDIM * 2);
    bf16_t* inpb   = (bf16_t*)alloc((size_t)BATCH * SLEN * 256 * 2);
    float*  gif    = (float*) alloc((size_t)BATCH * SLEN * 384 * 4);
    float*  ghf    = (float*) alloc((size_t)BATCH * SLEN * 384 * 4);
    bf16_t* finb   = (bf16_t*)alloc((size_t)BATCH * SLEN * HDIM * 2);
    bf16_t* lastsgb= (bf16_t*)alloc(BATCH * 256 * 2);
    float*  qf     = (float*) alloc(BATCH * HDIM * 4);
    float*  kvf    = (float*) alloc((size_t)BATCH * SLEN * 256 * 4);
    bf16_t* ctxb   = (bf16_t*)alloc(BATCH * HDIM * 2);
    bf16_t* fwb    = (bf16_t*)alloc((size_t)BATCH * SLEN * HDIM * 2);
    bf16_t* ub     = (bf16_t*)alloc((size_t)NCAND * 384 * 2);
    float*  extraf = (float*) alloc((size_t)BATCH * NCAND * 4);

    auto cvt = [&](const float* s, bf16_t* d, int n) {
        k_cvt<<<(n + 255) / 256, 256, 0, stream>>>(s, d, n);
    };

    // ---- Stage A: global graph conv ----
    k_zero<<<(NITEMS * HDIM + 255) / 256, 256, 0, stream>>>(aggf, NITEMS * HDIM);
    k_scatter<<<NEDGE, HDIM, 0, stream>>>(erow, ecol, ew, emb, aggf);
    cvt(emb, embb, NITEMS * HDIM);
    cvt(aggf, aggb, NITEMS * HDIM);
    cvt(gW, gWb, HDIM * HDIM);
    cvt(liW, liWb, HDIM * HDIM);
    cvt(loW, loWb, HDIM * HDIM);
    cvt(wih, wihb, 384 * 256);
    cvt(whh, whhb, 384 * 128);
    cvt(inW, inWb, 384 * 128);
    cvt(outW, outWb, HDIM * HDIM);
    cvt(wtW, wtWb, HDIM * HDIM);
    k_tcvt<<<(128 * 384 + 255) / 256, 256, 0, stream>>>(w3, w3tb, 128, 384);
    // hg = relu(agg @ gW^T + gb)
    gemm(stream, aggb, HDIM, gWb, HDIM, gb, hgf, nullptr, HDIM, NITEMS, HDIM, HDIM, 1);

    // ---- Stage B: session GNN (GRU cell) ----
    k_gather<<<BATCH * SLEN, HDIM, 0, stream>>>(items, emb, hgf, h0f, h0b, sgfb);
    gemm(stream, h0b, HDIM, liWb, HDIM, liB, nullptr, hinb, HDIM, BATCH * SLEN, HDIM, HDIM, 0);
    gemm(stream, h0b, HDIM, loWb, HDIM, loB, nullptr, houtb, HDIM, BATCH * SLEN, HDIM, HDIM, 0);
    k_adjmm<<<BATCH, 256, 0, stream>>>(adj, hinb, houtb, inpb);
    gemm(stream, inpb, 256, wihb, 256, bih, gif, nullptr, 384, BATCH * SLEN, 384, 256, 0);
    gemm(stream, h0b, HDIM, whhb, HDIM, bhh, ghf, nullptr, 384, BATCH * SLEN, 384, HDIM, 0);
    k_gru_final<<<BATCH * SLEN, HDIM, 0, stream>>>(gif, ghf, h0f, sgfb, pos, items, lens,
                                                   finb, lastsgb);

    // ---- Stage C: multi-head self-attention -> s_global ----
    gemm(stream, lastsgb, 256, inWb, HDIM, inB, qf, nullptr, HDIM, BATCH, HDIM, HDIM, 0);
    gemm(stream, finb, HDIM, inWb + HDIM * HDIM, HDIM, inB + HDIM, kvf, nullptr, 256,
         BATCH * SLEN, 256, HDIM, 0);
    k_attn<<<BATCH * NHEAD, 64, 0, stream>>>(qf, kvf, items, ctxb);
    gemm(stream, ctxb, HDIM, outWb, HDIM, outB, nullptr, lastsgb + HDIM, 256,
         BATCH, HDIM, HDIM, 0);

    // ---- Stage D: fused target-attention scoring ----
    // fw = final @ w_target^T ; u = cand @ w3 ; extra = [last|s_global] @ u[:,128:]^T
    gemm(stream, finb, HDIM, wtWb, HDIM, nullptr, nullptr, fwb, HDIM,
         BATCH * SLEN, HDIM, HDIM, 0);
    gemm(stream, embb + HDIM, HDIM, w3tb, HDIM, nullptr, nullptr, ub, 384,
         NCAND, 384, HDIM, 0);
    gemm(stream, lastsgb, 256, ub + HDIM, 384, nullptr, extraf, nullptr, NCAND,
         BATCH, NCAND, 256, 0);
    k_score<<<dim3((NCAND + 15) / 16, BATCH), 128, 0, stream>>>(
        fwb, finb, embb + HDIM, ub, extraf, items, out);
}
